// Model_75548474736945
// MI455X (gfx1250) — compile-verified, matrix operand order checked by reading, not run on hardware
//
#include <hip/hip_runtime.h>
#include <math.h>

// ---------------- model constants ----------------
#define BD   2
#define LE   2048
#define LD   1024
#define DM   256
#define HH   8
#define DH   32
#define DFF  1024
#define PRED 512
#define COUT 7
#define MARK 4

typedef _Float16 v8h  __attribute__((ext_vector_type(8)));
typedef _Float16 v16h __attribute__((ext_vector_type(16)));
typedef float    v8f  __attribute__((ext_vector_type(8)));

union Frag { v16h v; v8h h[2]; };

__device__ __forceinline__ v8f zero8f() {
    v8f z = {0.f,0.f,0.f,0.f,0.f,0.f,0.f,0.f};
    return z;
}

// =====================================================================
// f32 -> f16 conversion (weights / activations)
// =====================================================================
__global__ void cvt_f32_f16(const float* __restrict__ src,
                            _Float16* __restrict__ dst, int n) {
    int i = blockIdx.x * blockDim.x + threadIdx.x;
    if (i < n) dst[i] = (_Float16)src[i];
}

// =====================================================================
// Embedding: circular conv1d(k=3) + time-feature linear + sinusoidal PE
// =====================================================================
__global__ void embed_kernel(const float* __restrict__ X,     // [B,L,7]
                             const float* __restrict__ MK,    // [B,L,4]
                             const float* __restrict__ CW,    // [D,7,3]
                             const float* __restrict__ TW,    // [D,4]
                             float* __restrict__ outF,        // [B,L,D]
                             _Float16* __restrict__ outH,     // [B,L,D]
                             int L) {
    int idx = blockIdx.x * blockDim.x + threadIdx.x;
    int total = BD * L * DM;
    if (idx >= total) return;
    int d = idx % DM;
    int l = (idx / DM) % L;
    int b = idx / (DM * L);

    float acc = 0.f;
    #pragma unroll
    for (int t = 0; t < 3; ++t) {
        int ls = l + t - 1;
        ls = (ls + L) % L;                       // circular pad
        const float* xr = X + ((size_t)b * L + ls) * COUT;
        #pragma unroll
        for (int c = 0; c < COUT; ++c)
            acc += xr[c] * CW[(d * COUT + c) * 3 + t];
    }
    const float* mr = MK + ((size_t)b * L + l) * MARK;
    #pragma unroll
    for (int m = 0; m < MARK; ++m)
        acc += mr[m] * TW[d * MARK + m];

    int i2 = d & ~1;
    float dv  = __expf((float)i2 * (-9.210340371976184f / (float)DM)); // -ln(1e4)/D
    float ang = (float)l * dv;
    acc += (d & 1) ? __cosf(ang) : __sinf(ang);

    outF[idx] = acc;
    outH[idx] = (_Float16)acc;
}

// =====================================================================
// WMMA GEMM:  Y[M,N] = A[M,K] @ W[N,K]^T (+bias) (+gelu) (+resid)
// block = 128 threads (4 waves); wave w -> 16x64 strip; grid (M/64, N/64).
// The shared 64xK weight tile is streamed into LDS with double-buffered
// global_load_async_to_lds_b128 (ASYNCcnt pipe), so WMMAs wait only on
// s_wait_asynccnt<=2 for the previous stage while the next stage is in
// flight; B-fragments come from LDS, A-fragments from VMEM.
// =====================================================================
__global__ void gemm_wmma(const _Float16* __restrict__ A,
                          const _Float16* __restrict__ W,
                          const float* __restrict__ bias,
                          const float* __restrict__ resid,
                          float* __restrict__ outF,
                          _Float16* __restrict__ outH,
                          int M, int N, int K, int gelu) {
    __shared__ __align__(64) _Float16 WT[2][64 * 32];

    int tid  = threadIdx.x;
    int lane = tid & 31;
    int wave = tid >> 5;
    int m0   = blockIdx.x * 64 + wave * 16;
    int n0   = blockIdx.y * 64;
    int lm   = lane & 15;
    int kb   = (lane < 16) ? 0 : 8;

    // staging map: 64 rows x 64 bytes = 256 x 16B chunks, 2 per thread
    int c0 = tid,       r0 = c0 >> 2, o0 = (c0 & 3) * 8;
    int c1 = tid + 128, r1 = c1 >> 2, o1 = (c1 & 3) * 8;
    unsigned lds0[2], lds1[2];
    #pragma unroll
    for (int bf = 0; bf < 2; ++bf) {
        lds0[bf] = (unsigned)(uintptr_t)&WT[bf][r0 * 32 + o0];
        lds1[bf] = (unsigned)(uintptr_t)&WT[bf][r1 * 32 + o1];
    }

    v8f acc[4] = {zero8f(), zero8f(), zero8f(), zero8f()};

    const _Float16* Arow = A + (size_t)(m0 + lm) * K;
    const int nsteps = K >> 5;

    // stage 0
    {
        const void* g0 = (const void*)(W + (size_t)(n0 + r0) * K + o0);
        const void* g1 = (const void*)(W + (size_t)(n0 + r1) * K + o1);
        asm volatile("global_load_async_to_lds_b128 %0, %1, off"
                     :: "v"(lds0[0]), "v"(g0) : "memory");
        asm volatile("global_load_async_to_lds_b128 %0, %1, off"
                     :: "v"(lds1[0]), "v"(g1) : "memory");
    }

    for (int s = 0; s < nsteps; ++s) {
        int k   = s << 5;
        int buf = s & 1;
        if (s + 1 < nsteps) {
            // prefetch next stage into the other buffer (async pipe)
            const void* g0 = (const void*)(W + (size_t)(n0 + r0) * K + k + 32 + o0);
            const void* g1 = (const void*)(W + (size_t)(n0 + r1) * K + k + 32 + o1);
            asm volatile("global_load_async_to_lds_b128 %0, %1, off"
                         :: "v"(lds0[buf ^ 1]), "v"(g0) : "memory");
            asm volatile("global_load_async_to_lds_b128 %0, %1, off"
                         :: "v"(lds1[buf ^ 1]), "v"(g1) : "memory");
            asm volatile("s_wait_asynccnt 0x2" ::: "memory");  // stage s done
        } else {
            asm volatile("s_wait_asynccnt 0x0" ::: "memory");
        }
        __syncthreads();   // all waves' chunks of stage s visible

        if (k + 128 < K) {
            __builtin_prefetch(Arow + k + 128, 0, 1);  // global_prefetch_b8
        }
        Frag a;
        a.h[0] = *(const v8h*)(Arow + k + kb);
        a.h[1] = *(const v8h*)(Arow + k + 16 + kb);
        #pragma unroll
        for (int j = 0; j < 4; ++j) {
            const _Float16* wl = &WT[buf][(j * 16 + lm) * 32];
            Frag bf;
            bf.h[0] = *(const v8h*)(wl + kb);       // ds_load_b128
            bf.h[1] = *(const v8h*)(wl + 16 + kb);
            acc[j] = __builtin_amdgcn_wmma_f32_16x16x32_f16(
                false, a.v, false, bf.v, (short)0, acc[j], false, false);
        }
        __syncthreads();   // reads of buf done before next stage overwrites it
    }

    int hi = (lane < 16) ? 0 : 8;
    #pragma unroll
    for (int j = 0; j < 4; ++j) {
        int n = n0 + j * 16 + lm;
        float bv = bias ? bias[n] : 0.f;
        #pragma unroll
        for (int r = 0; r < 8; ++r) {
            int m = m0 + r + hi;
            float v = acc[j][r] + bv;
            if (gelu) v = 0.5f * v * (1.f + erff(v * 0.70710678118654752f));
            if (resid) v += resid[(size_t)m * N + n];
            if (outF) outF[(size_t)m * N + n] = v;
            if (outH) outH[(size_t)m * N + n] = (_Float16)v;
        }
    }
}

// =====================================================================
// V transpose: [B,L,D] f16 -> [B,H,dh,L] f16
// =====================================================================
__global__ void vtrans_kernel(const _Float16* __restrict__ V,
                              _Float16* __restrict__ Vt, int L) {
    int idx = blockIdx.x * blockDim.x + threadIdx.x;
    int total = BD * L * DM;
    if (idx >= total) return;
    int d = idx % DM;
    int l = (idx / DM) % L;
    int b = idx / (DM * L);
    int h = d >> 5, c = d & 31;
    Vt[((size_t)((b * HH + h) * DH + c)) * L + l] = V[idx];
}

// =====================================================================
// Flash attention, one wave (32 threads) per (b, h, 16-query tile).
// =====================================================================
__global__ void attn_flash(const _Float16* __restrict__ Q,   // [B,Lq,D]
                           const _Float16* __restrict__ Kh,  // [B,Lk,D]
                           const _Float16* __restrict__ Vt,  // [B,H,dh,Lk]
                           _Float16* __restrict__ O,         // [B,Lq,D]
                           int Lq, int Lk, int causal) {
    __shared__ __align__(16) _Float16 P[16 * 32];

    int lane = threadIdx.x & 31;
    int lm   = lane & 15;
    int kb   = (lane < 16) ? 0 : 8;
    int hi   = (lane < 16) ? 0 : 8;
    int bh   = blockIdx.x;
    int b = bh / HH, h = bh % HH;
    int qt = blockIdx.y;

    const _Float16* Qrow = Q + ((size_t)(b * Lq + qt * 16 + lm)) * DM + h * DH;
    Frag qa;
    qa.h[0] = *(const v8h*)(Qrow + kb);
    qa.h[1] = *(const v8h*)(Qrow + 16 + kb);

    float mrun[8], lrun[8];
    #pragma unroll
    for (int r = 0; r < 8; ++r) { mrun[r] = -1e30f; lrun[r] = 0.f; }
    v8f o0 = zero8f(), o1 = zero8f();

    const float scale = 0.1767766952966369f;   // 1/sqrt(32)
    int kend = causal ? (((qt * 16 + 16) + 31) / 32) * 32 : Lk;
    if (kend > Lk) kend = Lk;

    for (int kc = 0; kc < kend; kc += 32) {
        Frag k0, k1;
        {
            const _Float16* Kr0 = Kh + ((size_t)(b * Lk + kc + lm)) * DM + h * DH;
            const _Float16* Kr1 = Kh + ((size_t)(b * Lk + kc + 16 + lm)) * DM + h * DH;
            k0.h[0] = *(const v8h*)(Kr0 + kb);  k0.h[1] = *(const v8h*)(Kr0 + 16 + kb);
            k1.h[0] = *(const v8h*)(Kr1 + kb);  k1.h[1] = *(const v8h*)(Kr1 + 16 + kb);
        }
        v8f s0 = zero8f(), s1 = zero8f();
        s0 = __builtin_amdgcn_wmma_f32_16x16x32_f16(false, qa.v, false, k0.v,
                                                    (short)0, s0, false, false);
        s1 = __builtin_amdgcn_wmma_f32_16x16x32_f16(false, qa.v, false, k1.v,
                                                    (short)0, s1, false, false);

        int col0 = kc + lm, col1 = col0 + 16;
        #pragma unroll
        for (int r = 0; r < 8; ++r) {
            int row = qt * 16 + r + hi;
            float t0 = s0[r] * scale;
            float t1 = s1[r] * scale;
            if (causal) {
                if (col0 > row) t0 = -1e30f;
                if (col1 > row) t1 = -1e30f;
            }
            float cm = fmaxf(t0, t1);
            #pragma unroll
            for (int d = 8; d >= 1; d >>= 1)
                cm = fmaxf(cm, __shfl_xor(cm, d, 32));
            float mnew = fmaxf(mrun[r], cm);
            float f = __expf(mrun[r] - mnew);
            mrun[r] = mnew;

            float e0 = __expf(t0 - mnew);
            float e1 = __expf(t1 - mnew);
            float rs = e0 + e1;
            #pragma unroll
            for (int d = 8; d >= 1; d >>= 1)
                rs += __shfl_xor(rs, d, 32);
            lrun[r] = lrun[r] * f + rs;
            o0[r] *= f;  o1[r] *= f;

            P[(r + hi) * 32 + lm]      = (_Float16)e0;
            P[(r + hi) * 32 + 16 + lm] = (_Float16)e1;
        }
        __syncthreads();

        Frag pa;
        pa.h[0] = *(const v8h*)&P[lm * 32 + kb];
        pa.h[1] = *(const v8h*)&P[lm * 32 + 16 + kb];
        Frag bv0, bv1;
        {
            const _Float16* Vr0 = Vt + ((size_t)((b * HH + h) * DH + lm)) * Lk + kc;
            const _Float16* Vr1 = Vt + ((size_t)((b * HH + h) * DH + 16 + lm)) * Lk + kc;
            bv0.h[0] = *(const v8h*)(Vr0 + kb);  bv0.h[1] = *(const v8h*)(Vr0 + 16 + kb);
            bv1.h[0] = *(const v8h*)(Vr1 + kb);  bv1.h[1] = *(const v8h*)(Vr1 + 16 + kb);
        }
        o0 = __builtin_amdgcn_wmma_f32_16x16x32_f16(false, pa.v, false, bv0.v,
                                                    (short)0, o0, false, false);
        o1 = __builtin_amdgcn_wmma_f32_16x16x32_f16(false, pa.v, false, bv1.v,
                                                    (short)0, o1, false, false);
        __syncthreads();
    }

    #pragma unroll
    for (int r = 0; r < 8; ++r) {
        float inv = 1.f / lrun[r];
        int row = qt * 16 + r + hi;
        _Float16* orow = O + ((size_t)(b * Lq + row)) * DM + h * DH;
        orow[lm]      = (_Float16)(o0[r] * inv);
        orow[16 + lm] = (_Float16)(o1[r] * inv);
    }
}

// =====================================================================
// LayerNorm over last dim (256): 128-thread block = 4 waves = 4 rows.
// Optional residual add before stats; writes f32 and/or f16.
// =====================================================================
__global__ void ln_kernel(const float* __restrict__ X,
                          const float* __restrict__ R,
                          const float* __restrict__ g,
                          const float* __restrict__ bt,
                          float* __restrict__ outF,
                          _Float16* __restrict__ outH) {
    int row  = blockIdx.x * 4 + (threadIdx.x >> 5);
    int lane = threadIdx.x & 31;
    const float* xr = X + (size_t)row * DM;
    const float* rr = R ? R + (size_t)row * DM : nullptr;
    float v[8], s = 0.f, ss = 0.f;
    #pragma unroll
    for (int i = 0; i < 8; ++i) {
        int c = lane + 32 * i;
        float x = xr[c];
        if (rr) x += rr[c];
        v[i] = x; s += x; ss += x * x;
    }
    #pragma unroll
    for (int d = 16; d >= 1; d >>= 1) {
        s  += __shfl_xor(s,  d, 32);
        ss += __shfl_xor(ss, d, 32);
    }
    float mean = s * (1.f / DM);
    float var  = ss * (1.f / DM) - mean * mean;
    float rstd = rsqrtf(var + 1e-5f);
    #pragma unroll
    for (int i = 0; i < 8; ++i) {
        int c = lane + 32 * i;
        float y = (v[i] - mean) * rstd * g[c] + bt[c];
        if (outF) outF[(size_t)row * DM + c] = y;
        if (outH) outH[(size_t)row * DM + c] = (_Float16)y;
    }
}

// =====================================================================
// Final projection D->7 and PRED slice
// =====================================================================
__global__ void proj_kernel(const float* __restrict__ X,   // [B,LD,D]
                            const float* __restrict__ Pw,  // [7,D]
                            const float* __restrict__ Pb,  // [7]
                            float* __restrict__ out) {     // [B,PRED,7]
    int idx = blockIdx.x * blockDim.x + threadIdx.x;
    int total = BD * PRED * COUT;
    if (idx >= total) return;
    int c = idx % COUT;
    int p = (idx / COUT) % PRED;
    int b = idx / (COUT * PRED);
    int l = LD - PRED + p;
    const float* xr = X + ((size_t)b * LD + l) * DM;
    const float* wr = Pw + c * DM;
    float acc = Pb[c];
    for (int k = 0; k < DM; ++k) acc += xr[k] * wr[k];
    out[idx] = acc;
}

// =====================================================================
// Host orchestration
// =====================================================================
static inline size_t bump(size_t& off, size_t bytes) {
    size_t p = off;
    off = (off + bytes + 255) & ~(size_t)255;
    return p;
}

extern "C" void kernel_launch(void* const* d_in, const int* in_sizes, int n_in,
                              void* d_out, int out_size, void* d_ws, size_t ws_size,
                              hipStream_t stream) {
    (void)in_sizes; (void)n_in; (void)out_size; (void)ws_size;
    char* ws = (char*)d_ws;
    size_t off = 0;

    const size_t N_ENCW  = 2 * 4 * DM * DM;
    const size_t N_DECW  = 8 * DM * DM;
    const size_t N_ENCF1 = 2 * DFF * DM;
    const size_t N_ENCF2 = 2 * DM * DFF;
    const size_t N_DECF1 = DFF * DM;
    const size_t N_DECF2 = DM * DFF;
    _Float16* hENCW  = (_Float16*)(ws + bump(off, N_ENCW  * 2));
    _Float16* hDECW  = (_Float16*)(ws + bump(off, N_DECW  * 2));
    _Float16* hENCF1 = (_Float16*)(ws + bump(off, N_ENCF1 * 2));
    _Float16* hENCF2 = (_Float16*)(ws + bump(off, N_ENCF2 * 2));
    _Float16* hDECF1 = (_Float16*)(ws + bump(off, N_DECF1 * 2));
    _Float16* hDECF2 = (_Float16*)(ws + bump(off, N_DECF2 * 2));

    const size_t MAXM = (size_t)BD * LE;            // 4096
    const size_t MDEC = (size_t)BD * LD;            // 2048
    _Float16* x16   = (_Float16*)(ws + bump(off, MAXM * DM * 2));
    _Float16* q16   = (_Float16*)(ws + bump(off, MAXM * DM * 2));
    _Float16* k16   = (_Float16*)(ws + bump(off, MAXM * DM * 2));
    _Float16* v16   = (_Float16*)(ws + bump(off, MAXM * DM * 2));
    _Float16* vt16  = (_Float16*)(ws + bump(off, MAXM * DM * 2));
    _Float16* ao16  = (_Float16*)(ws + bump(off, MAXM * DM * 2));
    _Float16* y16   = (_Float16*)(ws + bump(off, MAXM * DFF * 2));
    _Float16* xd16  = (_Float16*)(ws + bump(off, MDEC * DM * 2));
    _Float16* enc16 = (_Float16*)(ws + bump(off, MAXM * DM * 2));
    float* x32   = (float*)(ws + bump(off, MAXM * DM * 4));
    float* xd32  = (float*)(ws + bump(off, MDEC * DM * 4));
    float* tmp32 = (float*)(ws + bump(off, MAXM * DM * 4));
    float* fin32 = (float*)(ws + bump(off, MDEC * DM * 4));

    const float* x_enc  = (const float*)d_in[0];
    const float* xm_enc = (const float*)d_in[1];
    const float* x_dec  = (const float*)d_in[2];
    const float* xm_dec = (const float*)d_in[3];
    const float* cw_enc = (const float*)d_in[4];
    const float* tw_enc = (const float*)d_in[5];
    const float* cw_dec = (const float*)d_in[6];
    const float* tw_dec = (const float*)d_in[7];
    const float* eW  = (const float*)d_in[8];
    const float* eB  = (const float*)d_in[9];
    const float* eF1 = (const float*)d_in[10];
    const float* eB1 = (const float*)d_in[11];
    const float* eF2 = (const float*)d_in[12];
    const float* eB2 = (const float*)d_in[13];
    const float* eNG = (const float*)d_in[14];
    const float* eNB = (const float*)d_in[15];
    const float* eFG = (const float*)d_in[16];
    const float* eFB = (const float*)d_in[17];
    const float* dW  = (const float*)d_in[18];
    const float* dB  = (const float*)d_in[19];
    const float* dF1 = (const float*)d_in[20];
    const float* dB1 = (const float*)d_in[21];
    const float* dF2 = (const float*)d_in[22];
    const float* dB2 = (const float*)d_in[23];
    const float* dNG = (const float*)d_in[24];
    const float* dNB = (const float*)d_in[25];
    const float* dFG = (const float*)d_in[26];
    const float* dFB = (const float*)d_in[27];
    const float* pW  = (const float*)d_in[28];
    const float* pB  = (const float*)d_in[29];

    auto cvt = [&](const float* s, _Float16* dst, size_t n) {
        cvt_f32_f16<<<dim3((unsigned)((n + 255) / 256)), dim3(256), 0, stream>>>(s, dst, (int)n);
    };
    cvt(eW,  hENCW,  N_ENCW);
    cvt(dW,  hDECW,  N_DECW);
    cvt(eF1, hENCF1, N_ENCF1);
    cvt(eF2, hENCF2, N_ENCF2);
    cvt(dF1, hDECF1, N_DECF1);
    cvt(dF2, hDECF2, N_DECF2);

    auto gemm = [&](const _Float16* A, const _Float16* W, const float* bias,
                    const float* resid, float* oF, _Float16* oH,
                    int M, int N, int K, int gelu) {
        gemm_wmma<<<dim3(M / 64, N / 64), dim3(128), 0, stream>>>(
            A, W, bias, resid, oF, oH, M, N, K, gelu);
    };
    auto ln = [&](const float* X, const float* R, const float* g, const float* b,
                  float* oF, _Float16* oH, size_t M) {
        ln_kernel<<<dim3((unsigned)(M / 4)), dim3(128), 0, stream>>>(X, R, g, b, oF, oH);
    };

    // =================== ENCODER ===================
    {
        int total = BD * LE * DM;
        embed_kernel<<<dim3((total + 255) / 256), dim3(256), 0, stream>>>(
            x_enc, xm_enc, cw_enc, tw_enc, x32, x16, LE);
    }
    for (int i = 0; i < 2; ++i) {
        const _Float16* Wq = hENCW + (size_t)(i * 4 + 0) * DM * DM;
        const _Float16* Wk = hENCW + (size_t)(i * 4 + 1) * DM * DM;
        const _Float16* Wv = hENCW + (size_t)(i * 4 + 2) * DM * DM;
        const _Float16* Wo = hENCW + (size_t)(i * 4 + 3) * DM * DM;
        const float* bq = eB + (size_t)(i * 4 + 0) * DM;
        const float* bk = eB + (size_t)(i * 4 + 1) * DM;
        const float* bv = eB + (size_t)(i * 4 + 2) * DM;
        const float* bo = eB + (size_t)(i * 4 + 3) * DM;

        gemm(x16, Wq, bq, nullptr, nullptr, q16, (int)MAXM, DM, DM, 0);
        gemm(x16, Wk, bk, nullptr, nullptr, k16, (int)MAXM, DM, DM, 0);
        gemm(x16, Wv, bv, nullptr, nullptr, v16, (int)MAXM, DM, DM, 0);
        {
            int total = BD * LE * DM;
            vtrans_kernel<<<dim3((total + 255) / 256), dim3(256), 0, stream>>>(v16, vt16, LE);
        }
        attn_flash<<<dim3(BD * HH, LE / 16), dim3(32), 0, stream>>>(
            q16, k16, vt16, ao16, LE, LE, 0);
        gemm(ao16, Wo, bo, x32, tmp32, nullptr, (int)MAXM, DM, DM, 0);
        ln(tmp32, nullptr, eNG + (size_t)(i * 2 + 0) * DM, eNB + (size_t)(i * 2 + 0) * DM,
           x32, x16, MAXM);

        gemm(x16, hENCF1 + (size_t)i * DFF * DM, eB1 + (size_t)i * DFF,
             nullptr, nullptr, y16, (int)MAXM, DFF, DM, 1);
        gemm(y16, hENCF2 + (size_t)i * DM * DFF, eB2 + (size_t)i * DM,
             x32, tmp32, nullptr, (int)MAXM, DM, DFF, 0);
        ln(tmp32, nullptr, eNG + (size_t)(i * 2 + 1) * DM, eNB + (size_t)(i * 2 + 1) * DM,
           x32, x16, MAXM);
    }
    ln(x32, nullptr, eFG, eFB, nullptr, enc16, MAXM);

    // =================== DECODER ===================
    {
        int total = BD * LD * DM;
        embed_kernel<<<dim3((total + 255) / 256), dim3(256), 0, stream>>>(
            x_dec, xm_dec, cw_dec, tw_dec, xd32, xd16, LD);
    }
    {
        // ---- causal self-attention ----
        gemm(xd16, hDECW + 0 * DM * DM, dB + 0 * DM, nullptr, nullptr, q16, (int)MDEC, DM, DM, 0);
        gemm(xd16, hDECW + 1 * DM * DM, dB + 1 * DM, nullptr, nullptr, k16, (int)MDEC, DM, DM, 0);
        gemm(xd16, hDECW + 2 * DM * DM, dB + 2 * DM, nullptr, nullptr, v16, (int)MDEC, DM, DM, 0);
        {
            int total = BD * LD * DM;
            vtrans_kernel<<<dim3((total + 255) / 256), dim3(256), 0, stream>>>(v16, vt16, LD);
        }
        attn_flash<<<dim3(BD * HH, LD / 16), dim3(32), 0, stream>>>(
            q16, k16, vt16, ao16, LD, LD, 1);
        gemm(ao16, hDECW + 3 * DM * DM, dB + 3 * DM, xd32, tmp32, nullptr, (int)MDEC, DM, DM, 0);
        ln(tmp32, nullptr, dNG + 0 * DM, dNB + 0 * DM, xd32, xd16, MDEC);

        // ---- cross-attention (K/V from encoder, Lk = 2048) ----
        gemm(xd16,  hDECW + 4 * DM * DM, dB + 4 * DM, nullptr, nullptr, q16, (int)MDEC, DM, DM, 0);
        gemm(enc16, hDECW + 5 * DM * DM, dB + 5 * DM, nullptr, nullptr, k16, (int)MAXM, DM, DM, 0);
        gemm(enc16, hDECW + 6 * DM * DM, dB + 6 * DM, nullptr, nullptr, v16, (int)MAXM, DM, DM, 0);
        {
            int total = BD * LE * DM;
            vtrans_kernel<<<dim3((total + 255) / 256), dim3(256), 0, stream>>>(v16, vt16, LE);
        }
        attn_flash<<<dim3(BD * HH, LD / 16), dim3(32), 0, stream>>>(
            q16, k16, vt16, ao16, LD, LE, 0);
        gemm(ao16, hDECW + 7 * DM * DM, dB + 7 * DM, xd32, tmp32, nullptr, (int)MDEC, DM, DM, 0);
        ln(tmp32, nullptr, dNG + 1 * DM, dNB + 1 * DM, xd32, xd16, MDEC);

        // ---- FFN ----
        gemm(xd16, hDECF1, dB1, nullptr, nullptr, y16, (int)MDEC, DFF, DM, 1);
        gemm(y16, hDECF2, dB2, xd32, tmp32, nullptr, (int)MDEC, DM, DFF, 0);
        ln(tmp32, nullptr, dNG + 2 * DM, dNB + 2 * DM, xd32, xd16, MDEC);
    }
    ln(xd32, nullptr, dFG, dFB, fin32, nullptr, MDEC);
    {
        int total = BD * PRED * COUT;
        proj_kernel<<<dim3((total + 255) / 256), dim3(256), 0, stream>>>(
            fin32, pW, pB, (float*)d_out);
    }
}
